// Attention_52140902973734
// MI455X (gfx1250) — matrix operand order repaired and checked by reading, not compiled, not run on hardware
//
#include <hip/hip_runtime.h>

// ---------------------------------------------------------------------------
// GPT-2 attention block for MI455X (gfx1250, wave32).
//   qkv = x @ w_attn + b_attn ; split heads ; present = stack(k,v)
//   a   = softmax(mask(q k^T / 8)) @ v ; out = a @ w_proj + b_proj
// d_out = [ out (4096*1024 f32) | present (2*2*16*2048*64 f32) ]
//
// CDNA5 paths used:
//   - v_wmma_f32_16x16x32_bf16 for both GEMM stages + attention
//   - TDM tensor_load_to_lds (inline asm, D# per ISA ch.8) double-buffered
//     with s_wait_tensorcnt for the GEMM global->LDS tiles
//   - packed hardware f32->bf16 conversion (v_cvt_pk_bf16_f32)
// ---------------------------------------------------------------------------

typedef __attribute__((ext_vector_type(16))) __bf16       bf16x16;
typedef __attribute__((ext_vector_type(2)))  __bf16       bf16x2;
typedef __attribute__((ext_vector_type(8)))  float        f32x8;
typedef __attribute__((ext_vector_type(4)))  unsigned int uint32x4;
typedef __attribute__((ext_vector_type(8)))  unsigned int uint32x8;

union FragBF {
    bf16x16        v;
    uint4          q[2];
    unsigned int   u[8];
    __bf16         b[16];
};

#define WMMA_BF16(A, B, C) \
    __builtin_amdgcn_wmma_f32_16x16x32_bf16(false, (A), false, (B), (short)0, (C), false, false)

// Packed f32 pair -> bf16 pair (one v_cvt_pk_bf16_f32 when available).
__device__ __forceinline__ unsigned pk2(float x, float y) {
#if __has_builtin(__builtin_amdgcn_cvt_pk_bf16_f32)
    union { bf16x2 v; unsigned u; } t;
    t.v = __builtin_amdgcn_cvt_pk_bf16_f32(x, y);
    return t.u;
#else
    union { __bf16 b[2]; unsigned u; } t;
    t.b[0] = (__bf16)x; t.b[1] = (__bf16)y;
    return t.u;
#endif
}

__device__ __forceinline__ unsigned short bfbits(float f) {
    union { __bf16 b; unsigned short u; } x;
    x.b = (__bf16)f;
    return x.u;
}

// Build a bf16 fragment half from 16 consecutive f32 values (8 packed cvts).
__device__ __forceinline__ FragBF loadF32Frag(const float* p) {
    FragBF f;
    const float4* p4 = (const float4*)p;
    float4 a = p4[0], b = p4[1], c = p4[2], d = p4[3];
    f.u[0] = pk2(a.x, a.y); f.u[1] = pk2(a.z, a.w);
    f.u[2] = pk2(b.x, b.y); f.u[3] = pk2(b.z, b.w);
    f.u[4] = pk2(c.x, c.y); f.u[5] = pk2(c.z, c.w);
    f.u[6] = pk2(d.x, d.y); f.u[7] = pk2(d.z, d.w);
    return f;
}

// --- TDM: issue a 2D tile load (tile 32 x 128 bf16, row stride K elems) ----
// D# group0: count=1 | lds_addr | global_addr | type=2 (ISA 8.3)
// D# group1: data_size=2B, tensor_dim0=32, tensor_dim1=128,
//            tile_dim0=32, tile_dim1=128, tensor_dim0_stride=K (ISA 8.4)
__device__ __forceinline__ uint32x8 tdm_g1(unsigned K) {
    uint32x8 g1 = {};
    g1[0] = 1u << 16;                       // workgroup_mask=0, data_size=1 (2B)
    g1[1] = 32u << 16;                      // tensor_dim0[15:0] = 32
    g1[2] = (128u << 16);                   // dim0 hi=0 | tensor_dim1[15:0] = 128
    g1[3] = (32u << 16);                    // dim1 hi=0 | tile_dim0 = 32
    g1[4] = 128u;                           // tile_dim1 = 128, tile_dim2 = 0
    g1[5] = K;                              // tensor_dim0_stride lo32
    g1[6] = 0u;
    g1[7] = 0u;
    return g1;
}

__device__ __forceinline__ void tdm_load(unsigned lds_byte_addr,
                                         unsigned long long gaddr,
                                         uint32x8 g1) {
    uint32x4 g0;
    g0[0] = 1u;                             // count=1, user descriptor
    g0[1] = lds_byte_addr;
    g0[2] = (unsigned)gaddr;                // global_addr[31:0]
    g0[3] = ((unsigned)(gaddr >> 32) & 0x01FFFFFFu) | 0x80000000u; // addr[56:32]|type=2
    asm volatile("tensor_load_to_lds %0, %1" :: "s"(g0), "s"(g1) : "memory");
}

__device__ __forceinline__ unsigned lds_addr_of(const void* p) {
    return (unsigned)(unsigned long long)(uintptr_t)p;  // flat low 32 = LDS offset
}

// ---------------------------------------------------------------------------
// Prologue 1: elementwise f32 -> bf16 (x -> xbf)
// ---------------------------------------------------------------------------
__global__ __launch_bounds__(256) void cvt_bf16(const float* __restrict__ X,
                                                unsigned short* __restrict__ Y) {
    const int i = (blockIdx.x * 256 + threadIdx.x) * 4;
    const float4 f = *(const float4*)(X + i);
    uint2 o;
    o.x = pk2(f.x, f.y);
    o.y = pk2(f.z, f.w);
    *(uint2*)(Y + i) = o;
}

// ---------------------------------------------------------------------------
// Prologue 2: W[K][N] f32 -> WT[N][K] bf16 (32x32 LDS tile transpose,
// packed-pair dword stores on the way out)
// ---------------------------------------------------------------------------
__global__ __launch_bounds__(256) void transpose_bf16(const float* __restrict__ W,
                                                      unsigned short* __restrict__ WT,
                                                      int K, int N) {
    __shared__ float tile[32][33];
    const int bn = blockIdx.x * 32, bk = blockIdx.y * 32;
    const int tx = threadIdx.x & 31, ty = threadIdx.x >> 5;  // ty: 0..7
#pragma unroll
    for (int r = ty; r < 32; r += 8)
        tile[r][tx] = W[(size_t)(bk + r) * N + bn + tx];
    __syncthreads();
    const int kt = (threadIdx.x & 15) * 2;          // k pair base
    const int n0 = threadIdx.x >> 4;                // 0..15
#pragma unroll
    for (int n = n0; n < 32; n += 16)
        *(unsigned*)&WT[(size_t)(bn + n) * K + bk + kt] =
            pk2(tile[kt][n], tile[kt + 1][n]);
}

// ---------------------------------------------------------------------------
// Tiled GEMM: C[M,N] = A[M,K] @ B[K,N] + bias.  A: bf16 [M][K], B: bf16 [N][K]
// (pre-transposed).  256 threads = 8 waves; block tile 128x128; wave 64x32.
// Both input tiles are fetched by the Tensor Data Mover into double-buffered
// LDS; wave 0 issues the DMA and gates it with s_wait_tensorcnt so stage i+1
// DMA overlaps stage i compute.  Hot loop: ds_load_b128 frags + 8 WMMAs.
// MODE 0: epilogue scatters q->bf16 ws, k/v->f32 present region.
// MODE 1: plain f32 store.
// ---------------------------------------------------------------------------
template <int MODE>
__global__ __launch_bounds__(256) void gemm128(
    const unsigned short* __restrict__ Abf,   // [M][K] bf16
    const unsigned short* __restrict__ BT,    // [N][K] bf16
    const float* __restrict__ bias,
    float* __restrict__ outPlain, unsigned short* __restrict__ qws,
    float* __restrict__ presOut, int M, int N, int K)
{
    __shared__ __align__(16) unsigned short As[2][128 * 32];  // [row][k]
    __shared__ __align__(16) unsigned short Bs[2][128 * 32];  // [col][k]

    const int tid  = threadIdx.x;
    const int bm   = blockIdx.y * 128, bn = blockIdx.x * 128;
    const int wid  = tid >> 5, lane = tid & 31, lo = lane & 15, hi = lane >> 4;
    const int wm   = (wid & 1) * 64, wn = (wid >> 1) * 32;

    const uint32x8 g1 = tdm_g1((unsigned)K);
    const unsigned long long gA = (unsigned long long)(uintptr_t)Abf + (size_t)bm * K * 2;
    const unsigned long long gB = (unsigned long long)(uintptr_t)BT + (size_t)bn * K * 2;
    const unsigned ldsA[2] = { lds_addr_of(&As[0][0]), lds_addr_of(&As[1][0]) };
    const unsigned ldsB[2] = { lds_addr_of(&Bs[0][0]), lds_addr_of(&Bs[1][0]) };

    f32x8 acc[4][2] = {};
    const int nsteps = K >> 5;

    if (wid == 0) {                      // prime stage 0
        tdm_load(ldsA[0], gA, g1);
        tdm_load(ldsB[0], gB, g1);
    }

    for (int i = 0; i < nsteps; ++i) {
        const int cur = i & 1;
        if (wid == 0) {
            if (i + 1 < nsteps) {        // overlap next-stage DMA with compute
                const unsigned long long koff = (unsigned long long)(i + 1) * 64; // 32 elems * 2B
                tdm_load(ldsA[1 - cur], gA + koff, g1);
                tdm_load(ldsB[1 - cur], gB + koff, g1);
                __builtin_amdgcn_s_wait_tensorcnt(2);   // stage i complete
            } else {
                __builtin_amdgcn_s_wait_tensorcnt(0);
            }
        }
        __syncthreads();                 // publish buffer `cur`

        FragBF af[4], bfg[2];
#pragma unroll
        for (int a = 0; a < 4; ++a) {    // A frag: lane row, K hi*16..+15
            const int base = (wm + a * 16 + lo) * 32 + hi * 16;
            af[a].q[0] = *(const uint4*)&As[cur][base];
            af[a].q[1] = *(const uint4*)&As[cur][base + 8];
        }
#pragma unroll
        for (int j = 0; j < 2; ++j) {    // B frag: lane col, K hi*16..+15
            const int base = (wn + j * 16 + lo) * 32 + hi * 16;
            bfg[j].q[0] = *(const uint4*)&Bs[cur][base];
            bfg[j].q[1] = *(const uint4*)&Bs[cur][base + 8];
        }
#pragma unroll
        for (int a = 0; a < 4; ++a)
#pragma unroll
            for (int j = 0; j < 2; ++j)
                acc[a][j] = WMMA_BF16(af[a].v, bfg[j].v, acc[a][j]);
        __syncthreads();                 // retire reads of buffer `cur`
    }

    // --- epilogue: C layout row M = r + 8*hi, col N = lo ---
#pragma unroll
    for (int a = 0; a < 4; ++a) {
#pragma unroll
        for (int j = 0; j < 2; ++j) {
            const int gn = bn + wn + j * 16 + lo;
            const float bv = bias[gn];
#pragma unroll
            for (int r = 0; r < 8; ++r) {
                const int gm = bm + wm + a * 16 + r + hi * 8;
                const float val = acc[a][j][r] + bv;
                if (MODE == 1) {
                    outPlain[(size_t)gm * N + gn] = val;
                } else {
                    const int which = gn >> 10;          // 0=q 1=k 2=v
                    const int hh = (gn >> 6) & 15, dd = gn & 63;
                    const int b = gm >> 11, s = gm & 2047;
                    if (which == 0)
                        qws[((size_t)(b * 16 + hh) * 2048 + s) * 64 + dd] = bfbits(val);
                    else
                        presOut[((size_t)(((which - 1) * 2 + b) * 16 + hh) * 2048 + s) * 64 + dd] = val;
                }
            }
        }
    }
}

// ---------------------------------------------------------------------------
// Flash attention: one wave per 16-row Q tile, 32-key blocks, online softmax.
// QK^T and PV via WMMA; P transposed through per-wave LDS (per-wave LDS ops
// are in-order); V staged transposed into LDS with packed-pair dword stores.
// No cross-wave barriers (causal trip counts differ per wave).
// ---------------------------------------------------------------------------
__global__ __launch_bounds__(128) void attn_flash(
    const unsigned short* __restrict__ Q,   // [B*H][S][64] bf16
    const float* __restrict__ PR,           // present: [2][B][H][S][64] f32
    unsigned short* __restrict__ aout)      // [B][S][1024] bf16 (merged heads)
{
    __shared__ __align__(16) unsigned short Pt[4][16 * 32];  // per-wave P tile
    __shared__ __align__(16) unsigned short Vt[4][64 * 32];  // per-wave V^T [d][key]

    const int wid = threadIdx.x >> 5, lane = threadIdx.x & 31;
    const int lo = lane & 15, hi = lane >> 4;
    const int bh = blockIdx.x, b = bh >> 4, h = bh & 15;
    const int qBase = (blockIdx.y * 4 + wid) * 16;

    const unsigned short* qrow = Q + ((size_t)bh * 2048 + qBase + lo) * 64;
    FragBF qa[2];
#pragma unroll
    for (int dh = 0; dh < 2; ++dh) {
        const int base = dh * 32 + hi * 16;
        qa[dh].q[0] = *(const uint4*)(qrow + base);
        qa[dh].q[1] = *(const uint4*)(qrow + base + 8);
    }

    const float* kP = PR + (size_t)bh * 2048 * 64;
    const float* vP = PR + (size_t)((2 + b) * 16 + h) * 2048 * 64;

    f32x8 acc[4] = {};
    float m[8], l[8];
#pragma unroll
    for (int r = 0; r < 8; ++r) { m[r] = -1e30f; l[r] = 0.0f; }

    const int kEnd = qBase + 16;                      // causal limit (exclusive)
    for (int kb = 0; kb < kEnd; kb += 32) {
        // stage V^T: lane owns key pair (2*lo, 2*lo+1) and d-half hi*32..+32;
        // each store is one packed bf16 pair along the key axis (dword LDS store)
        {
            const float* rA = vP + (size_t)(kb + 2 * lo) * 64 + hi * 32;
            const float* rB = rA + 64;
#pragma unroll
            for (int e4 = 0; e4 < 8; ++e4) {
                const float4 fA = ((const float4*)rA)[e4];
                const float4 fB = ((const float4*)rB)[e4];
                const int d = hi * 32 + e4 * 4;
                *(unsigned*)&Vt[wid][(d + 0) * 32 + 2 * lo] = pk2(fA.x, fB.x);
                *(unsigned*)&Vt[wid][(d + 1) * 32 + 2 * lo] = pk2(fA.y, fB.y);
                *(unsigned*)&Vt[wid][(d + 2) * 32 + 2 * lo] = pk2(fA.z, fB.z);
                *(unsigned*)&Vt[wid][(d + 3) * 32 + 2 * lo] = pk2(fA.w, fB.w);
            }
        }

        // scores: two 16x16 tiles (keys kb..+15, kb+16..+31), K-dim = d = 64
        f32x8 s0 = {}, s1 = {};
        {
            FragBF kf;
            kf = loadF32Frag(kP + (size_t)(kb + lo) * 64 + hi * 16);
            s0 = WMMA_BF16(qa[0].v, kf.v, s0);
            kf = loadF32Frag(kP + (size_t)(kb + lo) * 64 + 32 + hi * 16);
            s0 = WMMA_BF16(qa[1].v, kf.v, s0);
            kf = loadF32Frag(kP + (size_t)(kb + 16 + lo) * 64 + hi * 16);
            s1 = WMMA_BF16(qa[0].v, kf.v, s1);
            kf = loadF32Frag(kP + (size_t)(kb + 16 + lo) * 64 + 32 + hi * 16);
            s1 = WMMA_BF16(qa[1].v, kf.v, s1);
        }

        // online softmax (row M = r + 8*hi lives across lanes lo=0..15)
#pragma unroll
        for (int r = 0; r < 8; ++r) {
            const int row = qBase + r + hi * 8;
            const int c0 = kb + lo;
            float v0 = (c0 <= row)      ? s0[r] * 0.125f : -1.0e10f;
            float v1 = (c0 + 16 <= row) ? s1[r] * 0.125f : -1.0e10f;
            float t = fmaxf(v0, v1);
#pragma unroll
            for (int off = 8; off > 0; off >>= 1) t = fmaxf(t, __shfl_xor(t, off, 16));
            const float mn   = fmaxf(m[r], t);
            const float corr = __expf(m[r] - mn);
            const float p0 = __expf(v0 - mn), p1 = __expf(v1 - mn);
            float rs = p0 + p1;
#pragma unroll
            for (int off = 8; off > 0; off >>= 1) rs += __shfl_xor(rs, off, 16);
            l[r] = l[r] * corr + rs;
            m[r] = mn;
#pragma unroll
            for (int c = 0; c < 4; ++c) acc[c][r] *= corr;
            const int prow = r + hi * 8;
            Pt[wid][prow * 32 + lo]      = bfbits(p0);
            Pt[wid][prow * 32 + 16 + lo] = bfbits(p1);
        }

        // P as A-fragment (per-wave LDS ops are in-order; waits auto-inserted)
        FragBF pf;
        {
            const unsigned short* pr = &Pt[wid][lo * 32];
            pf.q[0] = *(const uint4*)(pr + hi * 16);
            pf.q[1] = *(const uint4*)(pr + hi * 16 + 8);
        }
        // PV: 4 d-chunks of 16 cols, B-frags from V^T
#pragma unroll
        for (int c = 0; c < 4; ++c) {
            const unsigned short* vp = &Vt[wid][(c * 16 + lo) * 32 + hi * 16];
            FragBF vf;
            vf.q[0] = *(const uint4*)vp;
            vf.q[1] = *(const uint4*)(vp + 8);
            acc[c] = WMMA_BF16(pf.v, vf.v, acc[c]);
        }
    }

    // normalize and store merged-head output (bf16 ws, consumed by proj GEMM)
    float inv[8];
#pragma unroll
    for (int r = 0; r < 8; ++r) inv[r] = 1.0f / l[r];
#pragma unroll
    for (int c = 0; c < 4; ++c)
#pragma unroll
        for (int r = 0; r < 8; ++r) {
            const int row = qBase + r + hi * 8;
            const int col = h * 64 + c * 16 + lo;
            aout[((size_t)(b * 2048 + row)) * 1024 + col] = bfbits(acc[c][r] * inv[r]);
        }
}

// ---------------------------------------------------------------------------
extern "C" void kernel_launch(void* const* d_in, const int* in_sizes, int n_in,
                              void* d_out, int out_size, void* d_ws, size_t ws_size,
                              hipStream_t stream) {
    const float* x      = (const float*)d_in[0];   // [2,2048,1024]
    const float* w_attn = (const float*)d_in[1];   // [1024,3072]
    const float* b_attn = (const float*)d_in[2];   // [3072]
    const float* w_proj = (const float*)d_in[3];   // [1024,1024]
    const float* b_proj = (const float*)d_in[4];   // [1024]

    float* out  = (float*)d_out;                   // first 4096*1024 = a-output
    float* pres = out + 4194304;                   // present [2,2,16,2048,64]

    char* ws = (char*)d_ws;
    unsigned short* xbf = (unsigned short*)(ws);                   //  8 MB x bf16
    unsigned short* qws = (unsigned short*)(ws + (8u  << 20));     //  8 MB q bf16
    unsigned short* aws = (unsigned short*)(ws + (16u << 20));     //  8 MB a bf16
    unsigned short* wTa = (unsigned short*)(ws + (24u << 20));     //  6 MB w_attn^T bf16
    unsigned short* wTp = (unsigned short*)(ws + (30u << 20));     //  2 MB w_proj^T bf16

    // Prologue: one-time convert/transpose (~24 MB of traffic, ~1 us @ 23 TB/s)
    cvt_bf16<<<4096, 256, 0, stream>>>(x, xbf);
    transpose_bf16<<<dim3(96, 32), 256, 0, stream>>>(w_attn, wTa, 1024, 3072);
    transpose_bf16<<<dim3(32, 32), 256, 0, stream>>>(w_proj, wTp, 1024, 1024);

    // 1) QKV GEMM: 4096x3072x1024, scatter q->ws(bf16), k/v->present(f32)
    gemm128<0><<<dim3(24, 32), 256, 0, stream>>>(
        xbf, wTa, b_attn, nullptr, qws, pres, 4096, 3072, 1024);

    // 2) flash attention per (b,h), 4 waves x 16 q-rows per block
    attn_flash<<<dim3(32, 32), 128, 0, stream>>>(qws, pres, aws);

    // 3) proj GEMM: 4096x1024x1024, plain f32 store into d_out
    gemm128<1><<<dim3(8, 32), 256, 0, stream>>>(
        aws, wTp, b_proj, out, nullptr, nullptr, 4096, 1024, 1024);
}
